// GCN1_40905268527513
// MI455X (gfx1250) — matrix operand order repaired and checked by reading, not compiled, not run on hardware
//
#include <hip/hip_runtime.h>
#include <hip/hip_bf16.h>

typedef __attribute__((ext_vector_type(16))) _Float16 v16h;
typedef __attribute__((ext_vector_type(8)))  _Float16 v8h;
typedef __attribute__((ext_vector_type(8)))  float    v8f;

#define CDIV(a,b) (((a)+(b)-1)/(b))

// Problem constants (match reference setup_inputs)
constexpr int NN  = 100000;  // nodes (divisible by 16)
constexpr int CIN = 15;
constexpr int C1  = 150;
constexpr int C2  = 200;
constexpr int C3  = 400;
constexpr int C4  = 200;
// K padded to multiple of 32 (WMMA depth)
constexpr int KP0 = 32;    // CIN pad
constexpr int KP1 = 160;   // C1 as K
constexpr int KP2 = 224;   // C2 as K
constexpr int KP3 = 416;   // C3 as K
// Output strides: multiple of 16 (tight, used by scatter/BN kernels)
constexpr int NP1 = 160;   // C1 pad
constexpr int NP2 = 208;   // C2 pad
constexpr int NP3 = 400;   // C3 exact
constexpr int NP4 = 208;   // C4 pad
// B-operand padding: multiple of 64 (4 n-tiles per wave, zero-padded rows)
constexpr int NB1 = 192;
constexpr int NB2 = 256;
constexpr int NB3 = 448;
constexpr int NB4 = 256;
constexpr int STATS_LD = 512;

// ---------------------------------------------------------------- degree / dinv
__global__ void k_deg_init(float* deg, int n) {
    int i = blockIdx.x * blockDim.x + threadIdx.x;
    if (i < n) deg[i] = 1.0f;  // self-loop
}
__global__ void k_deg_edges(const int* dst, float* deg, int nE) {
    int e = blockIdx.x * blockDim.x + threadIdx.x;
    if (e < nE) atomicAdd(&deg[dst[e]], 1.0f);
}
__global__ void k_deg_finish(float* deg, int n) {
    int i = blockIdx.x * blockDim.x + threadIdx.x;
    if (i < n) deg[i] = rsqrtf(deg[i]);   // deg >= 1 always
}

// ------------------------------------------------------- f32 -> f16 pad convert
__global__ void k_convert_pad(const float* __restrict__ X, _Float16* __restrict__ Xh,
                              int M, int K, int Kpad) {
    size_t idx = (size_t)blockIdx.x * blockDim.x + threadIdx.x;
    size_t total = (size_t)M * Kpad;
    if (idx >= total) return;
    int i = (int)(idx / Kpad), k = (int)(idx % Kpad);
    float v = (k < K) ? X[(size_t)i * K + k] : 0.0f;
    Xh[idx] = (_Float16)v;
}

// Weight W[K][N] row-major -> transposed padded f16 Wt[Npad][Kpad]
__global__ void k_prep_weight(const float* __restrict__ W, _Float16* __restrict__ Wt,
                              int K, int N, int Kpad, int Npad) {
    int idx = blockIdx.x * blockDim.x + threadIdx.x;
    int total = Npad * Kpad;
    if (idx >= total) return;
    int n = idx / Kpad, k = idx % Kpad;
    float v = (n < N && k < K) ? W[(size_t)k * N + n] : 0.0f;
    Wt[idx] = (_Float16)v;
}

// ----------------------------------------------------------------- WMMA GEMM
// C[M][ldc](f32) = A[M][K](f16) * Bt[Nb][K](f16)^T
// K % 32 == 0, M % 16 == 0, Nb % 64 == 0, ldc % 16 == 0, Nout <= ldc <= Nb.
// Each wave computes a 16x64 tile (4 accumulators) reusing one A fragment per
// K-step: 1 A load + 4 B loads -> 4 WMMAs. Cols in [Nout, ldc) written as 0,
// cols >= ldc skipped.
__global__ void k_wmma_gemm(const _Float16* __restrict__ A,
                            const _Float16* __restrict__ Bt,
                            float* __restrict__ C,
                            int M, int Nout, int ldc, int Nb, int K) {
    int wave = blockIdx.x * (blockDim.x >> 5) + (threadIdx.x >> 5);
    int ntg  = Nb >> 6;                  // groups of 4 n-tiles
    int tm   = wave / ntg;
    int tg   = wave % ntg;
    if (tm * 16 >= M) return;            // wave-uniform: EXEC stays all-ones
    int lane = threadIdx.x & 31;
    int h    = lane >> 4;                // lane half
    int r    = lane & 15;

    const _Float16* arow = A  + (size_t)(tm * 16 + r) * K;
    const _Float16* brow = Bt + (size_t)(tg * 64 + r) * K;
    const size_t bts = (size_t)16 * K;   // B row-tile stride

    v8f acc0 = {}, acc1 = {}, acc2 = {}, acc3 = {};
    for (int k0 = 0; k0 < K; k0 += 32) {
        // A frag: lane(h,r) holds row m, K = k0+16h .. +15 (contiguous, matches B)
        v8h alo = *(const v8h*)(arow + k0 + 16 * h);
        v8h ahi = *(const v8h*)(arow + k0 + 16 * h + 8);
        v16h a;
#pragma unroll
        for (int i = 0; i < 8; ++i) { a[i] = alo[i]; a[8 + i] = ahi[i]; }
        // B frags: lane(h,r) holds column n, K = k0+16h .. +15 (contiguous in Bt row)
        v16h b0 = *(const v16h*)(brow + 0 * bts + k0 + 16 * h);
        v16h b1 = *(const v16h*)(brow + 1 * bts + k0 + 16 * h);
        v16h b2 = *(const v16h*)(brow + 2 * bts + k0 + 16 * h);
        v16h b3 = *(const v16h*)(brow + 3 * bts + k0 + 16 * h);
        acc0 = __builtin_amdgcn_wmma_f32_16x16x32_f16(false, a, false, b0, (short)0, acc0, false, false);
        acc1 = __builtin_amdgcn_wmma_f32_16x16x32_f16(false, a, false, b1, (short)0, acc1, false, false);
        acc2 = __builtin_amdgcn_wmma_f32_16x16x32_f16(false, a, false, b2, (short)0, acc2, false, false);
        acc3 = __builtin_amdgcn_wmma_f32_16x16x32_f16(false, a, false, b3, (short)0, acc3, false, false);
    }
    // Store: C/D layout -> VGPR v, lane half h: row = tm*16 + 8h + v, col = tile_n + r
    v8f accs[4] = {acc0, acc1, acc2, acc3};
#pragma unroll
    for (int j = 0; j < 4; ++j) {
        int col = tg * 64 + j * 16 + r;
        if (col < ldc) {
            bool live = (col < Nout);
#pragma unroll
            for (int v = 0; v < 8; ++v) {
                int row = tm * 16 + 8 * h + v;
                C[(size_t)row * ldc + col] = live ? accs[j][v] : 0.0f;
            }
        }
    }
}

// ----------------------------------------------------------- graph aggregation
// agg[i][c] = hw[i][c] * dinv[i]^2   (self-loop term; padding cols stay 0)
__global__ void k_selfloop_init(const float* __restrict__ hw,
                                const float* __restrict__ dinv,
                                float* __restrict__ agg, int M, int ld) {
    size_t idx = (size_t)blockIdx.x * blockDim.x + threadIdx.x;
    size_t total = (size_t)M * ld;
    if (idx >= total) return;
    int i = (int)(idx / ld);
    float di = dinv[i];
    agg[idx] = hw[idx] * di * di;
}

// One block processes epb edges; threads cover channels.
__global__ void k_edge_scatter(const float* __restrict__ hw,
                               float* __restrict__ agg,
                               const int* __restrict__ src,
                               const int* __restrict__ dst,
                               const float* __restrict__ dinv,
                               int nE, int ld, int epb) {
    int c = threadIdx.x;
    if (c >= ld) return;
    int e0 = blockIdx.x * epb;
    int e1 = min(nE, e0 + epb);
    for (int e = e0; e < e1; ++e) {
        int s = src[e], d = dst[e];
        float coef = dinv[s] * dinv[d];
        atomicAdd(&agg[(size_t)d * ld + c], hw[(size_t)s * ld + c] * coef);
    }
}

// ----------------------------------------------------------------- epilogues
// out_h[i][c] = relu(X[i][c] + b[c]) as f16, zero-padded to ld_out
__global__ void k_bias_relu_f16(const float* __restrict__ X, const float* __restrict__ b,
                                _Float16* __restrict__ out,
                                int M, int ld_in, int ncols, int ld_out) {
    size_t idx = (size_t)blockIdx.x * blockDim.x + threadIdx.x;
    size_t total = (size_t)M * ld_out;
    if (idx >= total) return;
    int i = (int)(idx / ld_out), c = (int)(idx % ld_out);
    float v = 0.0f;
    if (c < ncols) v = fmaxf(X[(size_t)i * ld_in + c] + b[c], 0.0f);
    out[idx] = (_Float16)v;
}

__global__ void k_zero_f32(float* p, int n) {
    int i = blockIdx.x * blockDim.x + threadIdx.x;
    if (i < n) p[i] = 0.0f;
}

// partial column sums / sumsq (coalesced: threads cover consecutive cols per row)
__global__ void k_bn_partial(const float* __restrict__ X, float* __restrict__ acc,
                             int M, int ld, int ncols, int rpb) {
    int r0 = blockIdx.x * rpb;
    int r1 = min(M, r0 + rpb);
    for (int c = threadIdx.x; c < ncols; c += blockDim.x) {
        float s = 0.0f, sq = 0.0f;
        for (int r = r0; r < r1; ++r) {
            float v = X[(size_t)r * ld + c];
            s += v; sq += v * v;
        }
        atomicAdd(&acc[c], s);
        atomicAdd(&acc[STATS_LD + c], sq);
    }
}

// acc[c] -> mean ; acc[STATS_LD+c] -> rstd
__global__ void k_bn_finalize(float* acc, int ncols, float invM) {
    int c = blockIdx.x * blockDim.x + threadIdx.x;
    if (c >= ncols) return;
    float mean = acc[c] * invM;
    float var  = acc[STATS_LD + c] * invM - mean * mean;
    acc[c] = mean;
    acc[STATS_LD + c] = rsqrtf(var + 1e-5f);
}

// out_h[i][c] = relu((X[i][c]-mean[c])*rstd[c]*g[c]+be[c]) as f16, zero-padded
__global__ void k_bn_relu_f16(const float* __restrict__ X, const float* __restrict__ st,
                              const float* __restrict__ g, const float* __restrict__ be,
                              _Float16* __restrict__ out,
                              int M, int ld_in, int ncols, int ld_out) {
    size_t idx = (size_t)blockIdx.x * blockDim.x + threadIdx.x;
    size_t total = (size_t)M * ld_out;
    if (idx >= total) return;
    int i = (int)(idx / ld_out), c = (int)(idx % ld_out);
    float v = 0.0f;
    if (c < ncols) {
        float x = X[(size_t)i * ld_in + c];
        v = fmaxf((x - st[c]) * st[STATS_LD + c] * g[c] + be[c], 0.0f);
    }
    out[idx] = (_Float16)v;
}

// final: out[i] = lb4 + sum_c relu(bn(z4[i][c])) * lw4[c]
__global__ void k_final(const float* __restrict__ z, const float* __restrict__ st,
                        const float* __restrict__ g, const float* __restrict__ be,
                        const float* __restrict__ lw4, const float* __restrict__ lb4,
                        float* __restrict__ out, int M, int ld, int ncols) {
    int i = blockIdx.x * blockDim.x + threadIdx.x;
    if (i >= M) return;
    float a = lb4[0];
    const float* row = z + (size_t)i * ld;
    for (int c = 0; c < ncols; ++c) {
        float v = (row[c] - st[c]) * st[STATS_LD + c] * g[c] + be[c];
        v = fmaxf(v, 0.0f);
        a += v * lw4[c];
    }
    out[i] = a;
}

// =============================================================== host launcher
extern "C" void kernel_launch(void* const* d_in, const int* in_sizes, int n_in,
                              void* d_out, int out_size, void* d_ws, size_t ws_size,
                              hipStream_t stream) {
    (void)n_in; (void)out_size; (void)ws_size;
    const float* x   = (const float*)d_in[0];
    const int*   ei  = (const int*)d_in[1];
    const float* W1  = (const float*)d_in[2];
    const float* b1  = (const float*)d_in[3];
    const float* W2  = (const float*)d_in[4];
    /* b2 (d_in[5]) cancels inside batchnorm */
    const float* g1  = (const float*)d_in[6];
    const float* be1 = (const float*)d_in[7];
    const float* lw2 = (const float*)d_in[8];
    /* lb2 (d_in[9]) cancels inside batchnorm */
    const float* g2  = (const float*)d_in[10];
    const float* be2 = (const float*)d_in[11];
    const float* lw3 = (const float*)d_in[12];
    /* lb3 (d_in[13]) cancels inside batchnorm */
    const float* g3  = (const float*)d_in[14];
    const float* be3 = (const float*)d_in[15];
    const float* lw4 = (const float*)d_in[16];
    const float* lb4 = (const float*)d_in[17];
    float* out = (float*)d_out;

    const int E = in_sizes[1] / 2;
    const int* e_src = ei;
    const int* e_dst = ei + E;

    // ---- bump allocator over d_ws
    char* wp = (char*)d_ws;
    auto alloc = [&](size_t bytes) -> void* {
        void* r = (void*)wp;
        wp += (bytes + 255) & ~(size_t)255;
        return r;
    };
    float*    dinv  = (float*)   alloc((size_t)NN * 4);
    _Float16* xh    = (_Float16*)alloc((size_t)NN * KP0 * 2);
    _Float16* W1t   = (_Float16*)alloc((size_t)NB1 * KP0 * 2);
    _Float16* W2t   = (_Float16*)alloc((size_t)NB2 * KP1 * 2);
    _Float16* W3t   = (_Float16*)alloc((size_t)NB3 * KP2 * 2);
    _Float16* W4t   = (_Float16*)alloc((size_t)NB4 * KP3 * 2);
    float*    stats = (float*)   alloc((size_t)2 * STATS_LD * 4);
    float*    bufA  = (float*)   alloc((size_t)NN * KP3 * 4);  // hw1 / hw2 / z3
    float*    bufB  = (float*)   alloc((size_t)NN * KP2 * 4);  // agg1 / agg2 / z4
    _Float16* hbufA = (_Float16*)alloc((size_t)NN * KP3 * 2);  // h1h / h3h
    _Float16* hbufB = (_Float16*)alloc((size_t)NN * KP2 * 2);  // h2h

    auto gemm = [&](const _Float16* A, const _Float16* Bt, float* C,
                    int Nout, int ldc, int Nb, int K) {
        int waves  = (NN / 16) * (Nb / 64);
        int blocks = CDIV(waves, 8);
        k_wmma_gemm<<<blocks, 256, 0, stream>>>(A, Bt, C, NN, Nout, ldc, Nb, K);
    };
    const int EPB = 16;  // edges per scatter block

    // degrees -> dinv
    k_deg_init  <<<CDIV(NN, 256), 256, 0, stream>>>(dinv, NN);
    k_deg_edges <<<CDIV(E, 256),  256, 0, stream>>>(e_dst, dinv, E);
    k_deg_finish<<<CDIV(NN, 256), 256, 0, stream>>>(dinv, NN);

    // operand prep
    k_convert_pad<<<CDIV((size_t)NN * KP0, 256), 256, 0, stream>>>(x, xh, NN, CIN, KP0);
    k_prep_weight<<<CDIV(NB1 * KP0, 256), 256, 0, stream>>>(W1,  W1t, CIN, C1, KP0, NB1);
    k_prep_weight<<<CDIV(NB2 * KP1, 256), 256, 0, stream>>>(W2,  W2t, C1,  C2, KP1, NB2);
    k_prep_weight<<<CDIV(NB3 * KP2, 256), 256, 0, stream>>>(lw2, W3t, C2,  C3, KP2, NB3);
    k_prep_weight<<<CDIV(NB4 * KP3, 256), 256, 0, stream>>>(lw3, W4t, C3,  C4, KP3, NB4);

    // --- GCN layer 1: hw1 = x@W1 ; agg ; relu(+b1) -> f16
    gemm(xh, W1t, bufA, C1, NP1, NB1, KP0);
    k_selfloop_init<<<CDIV((size_t)NN * NP1, 256), 256, 0, stream>>>(bufA, dinv, bufB, NN, NP1);
    k_edge_scatter <<<CDIV(E, EPB), 256, 0, stream>>>(bufA, bufB, e_src, e_dst, dinv, E, NP1, EPB);
    k_bias_relu_f16<<<CDIV((size_t)NN * KP1, 256), 256, 0, stream>>>(bufB, b1, hbufA, NN, NP1, C1, KP1);

    // --- GCN layer 2: hw2 = h1@W2 ; agg ; BN1 + relu -> f16
    gemm(hbufA, W2t, bufA, C2, NP2, NB2, KP1);
    k_selfloop_init<<<CDIV((size_t)NN * NP2, 256), 256, 0, stream>>>(bufA, dinv, bufB, NN, NP2);
    k_edge_scatter <<<CDIV(E, EPB), 256, 0, stream>>>(bufA, bufB, e_src, e_dst, dinv, E, NP2, EPB);
    k_zero_f32   <<<CDIV(2 * STATS_LD, 256), 256, 0, stream>>>(stats, 2 * STATS_LD);
    k_bn_partial <<<CDIV(NN, 512), 256, 0, stream>>>(bufB, stats, NN, NP2, C2, 512);
    k_bn_finalize<<<CDIV(C2, 256), 256, 0, stream>>>(stats, C2, 1.0f / NN);
    k_bn_relu_f16<<<CDIV((size_t)NN * KP2, 256), 256, 0, stream>>>(bufB, stats, g1, be1, hbufB, NN, NP2, C2, KP2);

    // --- MLP 200 -> 400 ; BN2 + relu -> f16
    gemm(hbufB, W3t, bufA, C3, NP3, NB3, KP2);
    k_zero_f32   <<<CDIV(2 * STATS_LD, 256), 256, 0, stream>>>(stats, 2 * STATS_LD);
    k_bn_partial <<<CDIV(NN, 512), 256, 0, stream>>>(bufA, stats, NN, NP3, C3, 512);
    k_bn_finalize<<<CDIV(C3, 256), 256, 0, stream>>>(stats, C3, 1.0f / NN);
    k_bn_relu_f16<<<CDIV((size_t)NN * KP3, 256), 256, 0, stream>>>(bufA, stats, g2, be2, hbufA, NN, NP3, C3, KP3);

    // --- MLP 400 -> 200 ; BN3 + relu fused into final mat-vec
    gemm(hbufA, W4t, bufB, C4, NP4, NB4, KP3);
    k_zero_f32   <<<CDIV(2 * STATS_LD, 256), 256, 0, stream>>>(stats, 2 * STATS_LD);
    k_bn_partial <<<CDIV(NN, 512), 256, 0, stream>>>(bufB, stats, NN, NP4, C4, 512);
    k_bn_finalize<<<CDIV(C4, 256), 256, 0, stream>>>(stats, C4, 1.0f / NN);
    k_final      <<<CDIV(NN, 256), 256, 0, stream>>>(bufB, stats, g3, be3, lw4, lb4, out, NN, NP4, C4);
}
